// _netG_58600533786759
// MI455X (gfx1250) — matrix operand order repaired and checked by reading, not compiled, not run on hardware
//
// MI455X (gfx1250) implementation of the netG LSTM+attention+decoder reference.
//
// Strategy (reasoned from MI455X specs, no runtime measurement possible):
//  * Dominant cost: decoder [256,2048]x[2048,8964] GEMM each of 20 sequential
//    timesteps (~9.4 GFLOP/step of ~13 GFLOP/step total). All weights
//    (~55 MB bf16) + ctx embeddings fit in 192 MB L2 -> compute bound.
//  * Use v_wmma_f32_16x16x32_bf16 (wave32, f32 accum). Weights and the
//    time-invariant emb tensor are converted f32->bf16 ONCE per call into
//    workspace; per-step activations are cvt'd just before their GEMM, so the
//    hot GEMM loop is pure 16-bit b128 loads + WMMA.
//  * V=8964 is tile-ragged: instead of per-iteration column guards (which
//    cost s_and_saveexec + zero-fill movs around every WMMA in round 1), Wd2
//    is staged ZERO-PADDED to [9088,2048] and dec gets a padded ldc, so the
//    inner loop is guard-free; Nreal only gates the one-time bias read.
//  * Each wave computes TWO 16x16 tiles (block tile 16x128): one A fragment
//    feeds two independent WMMAs -> halves A traffic, doubles WMMA ILP.
//  * LSTM cell, attention (logits -> softmax -> weighted sum), and
//    log_softmax(BETA*dec) are bandwidth-light VALU/LDS-reduction kernels.

#include <hip/hip_runtime.h>
#include <hip/hip_bf16.h>
#include <math.h>

constexpr int B  = 256;
constexpr int T  = 20;
constexpr int H  = 512;
constexpr int NL = 2;
constexpr int V  = 8964;
constexpr int VP = 9088;          // V padded to multiple of 128
constexpr int LQ = 40;
constexpr int LH = 80;
constexpr int LI = 49;
constexpr float BETA_F = 3.0f;

typedef __bf16 bf16;
typedef __attribute__((ext_vector_type(16))) __bf16 v16bf;
typedef __attribute__((ext_vector_type(8)))  __bf16 bf16x8;
typedef __attribute__((ext_vector_type(8)))  float  v8f;

// ---------------------------------------------------------------------------
// Fragment loaders (CDNA5 ISA 7.12.2 wave32 layouts)
// A 16x32 bf16: laneLo = row M, laneHi selects K-halves:
//   elem e<8  -> K = 8*laneHi + e ;  elem e>=8 -> K = 8*laneHi + 16 + (e-8)
// B 32x16 bf16: laneLo = col N, same K order as A
// ---------------------------------------------------------------------------
__device__ __forceinline__ v16bf load_a(const bf16* __restrict__ p, int laneHi) {
    const bf16x8 lo = *(const bf16x8*)(p + 8 * laneHi);
    const bf16x8 hi = *(const bf16x8*)(p + 8 * laneHi + 16);
    v16bf a;
#pragma unroll
    for (int e = 0; e < 8; ++e) { a[e] = lo[e]; a[e + 8] = hi[e]; }
    return a;
}

__device__ __forceinline__ v16bf load_a(const float* __restrict__ p, int laneHi) {
    const float* p0 = p + 8 * laneHi;
    v16bf a;
#pragma unroll
    for (int e = 0; e < 8; ++e) { a[e] = (bf16)p0[e]; a[e + 8] = (bf16)p0[16 + e]; }
    return a;
}

__device__ __forceinline__ v16bf load_b(const bf16* __restrict__ p, int laneHi) {
    const bf16x8 lo = *(const bf16x8*)(p + 8 * laneHi);
    const bf16x8 hi = *(const bf16x8*)(p + 8 * laneHi + 16);
    v16bf b;
#pragma unroll
    for (int e = 0; e < 8; ++e) { b[e] = lo[e]; b[e + 8] = hi[e]; }
    return b;
}

// ---------------------------------------------------------------------------
// C[M,N] = act( A[M,K] @ W[N,K]^T + bias0 + bias1 (+ C if ACC) )
// Requirements: M % 16 == 0, N % 128 == 0, K % 32 == 0, W zero-padded to N
// rows, C has ldc >= N. flags: bit0 = accumulate; bits[2:1]: 0=none 1=tanh
// 2=relu. Nreal gates only the bias reads (bias arrays are unpadded).
// Block = 128 threads = 4 waves; block tile 16(M) x 128(N); each wave owns
// two adjacent 16x16 tiles sharing one A fragment.
// ---------------------------------------------------------------------------
template <typename AT>
__global__ __launch_bounds__(128) void gemm_wmma_nt(
    const AT* __restrict__ A, int lda,
    const bf16* __restrict__ W,           // [N,K] row-major (ldw = K), padded
    const float* __restrict__ bias0,
    const float* __restrict__ bias1,
    float* __restrict__ C, int ldc,
    int M, int N, int K, int Nreal, int flags)
{
    const int wave   = threadIdx.x >> 5;
    const int lane   = threadIdx.x & 31;
    const int laneHi = lane >> 4;
    const int laneLo = lane & 15;
    const int row0   = blockIdx.y * 16;
    const int col0   = (blockIdx.x * 8 + wave * 2) * 16;
    const int n0     = col0 + laneLo;
    const int n1     = n0 + 16;

    v8f acc0, acc1;
    if (flags & 1) {
#pragma unroll
        for (int r = 0; r < 8; ++r) {
            int m = row0 + r + 8 * laneHi;
            acc0[r] = C[(size_t)m * ldc + n0];
            acc1[r] = C[(size_t)m * ldc + n1];
        }
    } else {
#pragma unroll
        for (int r = 0; r < 8; ++r) { acc0[r] = 0.0f; acc1[r] = 0.0f; }
    }

    const AT*   arow  = A + (size_t)(row0 + laneLo) * lda;
    const bf16* wrow0 = W + (size_t)n0 * K;
    const bf16* wrow1 = W + (size_t)n1 * K;

    for (int k0 = 0; k0 < K; k0 += 32) {
        v16bf a  = load_a(arow + k0, laneHi);
        v16bf b0 = load_b(wrow0 + k0, laneHi);
        v16bf b1 = load_b(wrow1 + k0, laneHi);
        acc0 = __builtin_amdgcn_wmma_f32_16x16x32_bf16(
            false, a, false, b0, (short)0, acc0, false, false);
        acc1 = __builtin_amdgcn_wmma_f32_16x16x32_bf16(
            false, a, false, b1, (short)0, acc1, false, false);
    }

    const int act = (flags >> 1) & 3;
    float bia0 = 0.0f, bia1 = 0.0f;
    if (bias0) {
        if (n0 < Nreal) bia0 += bias0[n0];
        if (n1 < Nreal) bia1 += bias0[n1];
    }
    if (bias1) {
        if (n0 < Nreal) bia0 += bias1[n0];
        if (n1 < Nreal) bia1 += bias1[n1];
    }
#pragma unroll
    for (int r = 0; r < 8; ++r) {
        int m = row0 + r + 8 * laneHi;
        float v0 = acc0[r] + bia0;
        float v1 = acc1[r] + bia1;
        if (act == 1)      { v0 = tanhf(v0);       v1 = tanhf(v1); }
        else if (act == 2) { v0 = fmaxf(v0, 0.0f); v1 = fmaxf(v1, 0.0f); }
        C[(size_t)m * ldc + n0] = v0;
        C[(size_t)m * ldc + n1] = v1;
    }
}

// ---------------------------------------------------------------------------
// Elementwise / reduction kernels
// ---------------------------------------------------------------------------
__global__ __launch_bounds__(256) void cvt_f32_bf16_k(const float* __restrict__ src,
                                                      bf16* __restrict__ dst, size_t n) {
    size_t i = (size_t)blockIdx.x * blockDim.x + threadIdx.x;
    size_t stride = (size_t)gridDim.x * blockDim.x;
    for (; i < n; i += stride) dst[i] = (bf16)src[i];
}

// dst[Npad,K] = src[N,K] with rows >= N zero-filled
__global__ __launch_bounds__(256) void cvt_pad_rows_k(const float* __restrict__ src,
                                                      bf16* __restrict__ dst,
                                                      int N, int Npad, int K) {
    size_t total = (size_t)Npad * K;
    size_t stride = (size_t)gridDim.x * blockDim.x;
    for (size_t i = (size_t)blockIdx.x * blockDim.x + threadIdx.x; i < total;
         i += stride) {
        size_t row = i / (size_t)K;
        dst[i] = (row < (size_t)N) ? (bf16)src[i] : (bf16)0.0f;
    }
}

__global__ __launch_bounds__(256) void copy2d_k(const float* __restrict__ src, int lds_,
                                                float* __restrict__ dst, int ldd,
                                                int rows, int cols) {
    int total = rows * cols;
    for (int i = blockIdx.x * blockDim.x + threadIdx.x; i < total;
         i += gridDim.x * blockDim.x) {
        int r = i / cols, c = i % cols;
        dst[(size_t)r * ldd + c] = src[(size_t)r * lds_ + c];
    }
}

// gates [B,4H] in torch order i,f,g,o ; c,h are [B,H] slices (updated in place)
__global__ __launch_bounds__(256) void lstm_cell_k(const float* __restrict__ gates,
                                                   float* __restrict__ c,
                                                   float* __restrict__ h) {
    int i = blockIdx.x * 256 + threadIdx.x;
    if (i >= B * H) return;
    int b = i >> 9;          // /H (H=512)
    int hh = i & (H - 1);
    const float* g = gates + (size_t)b * (4 * H);
    float gi = g[hh], gf = g[H + hh], gg = g[2 * H + hh], go = g[3 * H + hh];
    float si = 1.0f / (1.0f + expf(-gi));
    float sf = 1.0f / (1.0f + expf(-gf));
    float so = 1.0f / (1.0f + expf(-go));
    float cn = sf * c[i] + si * tanhf(gg);
    c[i] = cn;
    h[i] = so * tanhf(cn);
}

// logits[b,l] = sum_h tanh(ctx[b,l,h] + proj[b,h]) * wa[h] + ba
__global__ __launch_bounds__(256) void attn_logits_k(const float* __restrict__ ctx,
                                                     const float* __restrict__ proj,
                                                     const float* __restrict__ wa,
                                                     const float* __restrict__ ba,
                                                     float* __restrict__ logits, int Lx) {
    int b = blockIdx.y, l = blockIdx.x;
    const float* c = ctx + ((size_t)b * Lx + l) * H;
    const float* p = proj + (size_t)b * H;
    float s = 0.0f;
    for (int h = threadIdx.x; h < H; h += 256)
        s += tanhf(c[h] + p[h]) * wa[h];
    __shared__ float red[256];
    red[threadIdx.x] = s;
    __syncthreads();
    for (int off = 128; off > 0; off >>= 1) {
        if (threadIdx.x < off) red[threadIdx.x] += red[threadIdx.x + off];
        __syncthreads();
    }
    if (threadIdx.x == 0) logits[(size_t)b * Lx + l] = red[0] + ba[0];
}

// out[b, :] = softmax(logits[b,:]) @ ctx[b]  (written with row stride ldo)
__global__ __launch_bounds__(256) void attn_combine_k(const float* __restrict__ ctx,
                                                      const float* __restrict__ logits,
                                                      float* __restrict__ outp, int ldo,
                                                      int Lx) {
    int b = blockIdx.x;
    int tid = threadIdx.x;
    __shared__ float w[128];
    if (tid < Lx) w[tid] = logits[(size_t)b * Lx + tid];
    __syncthreads();
    if (tid == 0) {
        float m = -3.4e38f;
        for (int l = 0; l < Lx; ++l) m = fmaxf(m, w[l]);
        float s = 0.0f;
        for (int l = 0; l < Lx; ++l) { w[l] = expf(w[l] - m); s += w[l]; }
        float inv = 1.0f / s;
        for (int l = 0; l < Lx; ++l) w[l] *= inv;
    }
    __syncthreads();
    const float* c0 = ctx + (size_t)b * Lx * H;
    for (int h = tid; h < H; h += 256) {
        float acc = 0.0f;
        for (int l = 0; l < Lx; ++l) acc += w[l] * c0[(size_t)l * H + h];
        outp[(size_t)b * ldo + h] = acc;
    }
}

// out[b,t,:] = log_softmax(BETA * dec[b,:]) ; dec has leading dim lddec
__global__ __launch_bounds__(256) void logsoftmax_k(const float* __restrict__ dec,
                                                    int lddec,
                                                    float* __restrict__ out, int t) {
    int b = blockIdx.x;
    const float* d = dec + (size_t)b * lddec;
    float* o = out + ((size_t)b * T + t) * (size_t)V;
    __shared__ float red[256];
    float m = -3.4e38f;
    for (int v = threadIdx.x; v < V; v += 256) m = fmaxf(m, BETA_F * d[v]);
    red[threadIdx.x] = m;
    __syncthreads();
    for (int off = 128; off > 0; off >>= 1) {
        if (threadIdx.x < off)
            red[threadIdx.x] = fmaxf(red[threadIdx.x], red[threadIdx.x + off]);
        __syncthreads();
    }
    m = red[0];
    __syncthreads();
    float s = 0.0f;
    for (int v = threadIdx.x; v < V; v += 256) s += expf(BETA_F * d[v] - m);
    red[threadIdx.x] = s;
    __syncthreads();
    for (int off = 128; off > 0; off >>= 1) {
        if (threadIdx.x < off) red[threadIdx.x] += red[threadIdx.x + off];
        __syncthreads();
    }
    float lse = m + logf(red[0]);
    for (int v = threadIdx.x; v < V; v += 256) o[v] = BETA_F * d[v] - lse;
}

// ---------------------------------------------------------------------------
// Host side
// ---------------------------------------------------------------------------
static inline void launch_gemm_bf(const bf16* A, int lda, const bf16* W,
                                  const float* b0, const float* b1, float* C, int ldc,
                                  int M, int N, int K, int Nreal, int flags,
                                  hipStream_t s) {
    dim3 grid(N / 128, M / 16);
    gemm_wmma_nt<bf16><<<grid, 128, 0, s>>>(A, lda, W, b0, b1, C, ldc,
                                            M, N, K, Nreal, flags);
}
static inline void launch_gemm_f32(const float* A, int lda, const bf16* W,
                                   const float* b0, const float* b1, float* C, int ldc,
                                   int M, int N, int K, int Nreal, int flags,
                                   hipStream_t s) {
    dim3 grid(N / 128, M / 16);
    gemm_wmma_nt<float><<<grid, 128, 0, s>>>(A, lda, W, b0, b1, C, ldc,
                                             M, N, K, Nreal, flags);
}
static inline void launch_cvt(const float* src, bf16* dst, size_t n, hipStream_t s) {
    cvt_f32_bf16_k<<<2048, 256, 0, s>>>(src, dst, n);
}

extern "C" void kernel_launch(void* const* d_in, const int* in_sizes, int n_in,
                              void* d_out, int out_size, void* d_ws, size_t ws_size,
                              hipStream_t stream) {
    (void)in_sizes; (void)n_in; (void)out_size; (void)ws_size;

    // ---- inputs (setup_inputs dict order) ----
    const float* emb      = (const float*)d_in[0];   // [B,T,H]
    const float* question = (const float*)d_in[1];   // [B,LQ,2H]
    const float* history  = (const float*)d_in[2];   // [B,LH,2H]
    const float* image    = (const float*)d_in[3];   // [B,LI,2H]
    const float* h0       = (const float*)d_in[4];   // [L,B,H]
    const float* c0       = (const float*)d_in[5];
    const float* Wih      = (const float*)d_in[6];   // [L,4H,H]
    const float* Whh      = (const float*)d_in[7];
    const float* bih      = (const float*)d_in[8];   // [L,4H]
    const float* bhh      = (const float*)d_in[9];
    const float* Wq   = (const float*)d_in[10]; const float* bq    = (const float*)d_in[11];
    const float* Wansq= (const float*)d_in[12]; const float* bansq = (const float*)d_in[13];
    const float* waq  = (const float*)d_in[14]; const float* baq   = (const float*)d_in[15];
    const float* Wh   = (const float*)d_in[16]; const float* bh    = (const float*)d_in[17];
    const float* Wansh= (const float*)d_in[18]; const float* bansh = (const float*)d_in[19];
    const float* wah  = (const float*)d_in[20]; const float* bah   = (const float*)d_in[21];
    const float* Wi   = (const float*)d_in[22]; const float* bi    = (const float*)d_in[23];
    const float* Wansi= (const float*)d_in[24]; const float* bansi = (const float*)d_in[25];
    const float* wai  = (const float*)d_in[26]; const float* bai   = (const float*)d_in[27];
    const float* Wcat = (const float*)d_in[28]; const float* bcat  = (const float*)d_in[29];
    const float* Wd1  = (const float*)d_in[30]; const float* bd1   = (const float*)d_in[31];
    const float* Wd2  = (const float*)d_in[32]; const float* bd2   = (const float*)d_in[33];

    float* out = (float*)d_out;

    // ---- workspace bump allocator ----
    char* base = (char*)d_ws;
    size_t off = 0;
    auto alloc = [&](size_t bytes) -> void* {
        off = (off + 255) & ~(size_t)255;
        void* p = base + off;
        off += bytes;
        return p;
    };
    auto allocF  = [&](size_t n) { return (float*)alloc(n * sizeof(float)); };
    auto allocBF = [&](size_t n) { return (bf16*)alloc(n * sizeof(bf16)); };

    // f32 activations
    float* q_emb  = allocF((size_t)B * LQ * H);
    float* h_emb  = allocF((size_t)B * LH * H);
    float* i_emb  = allocF((size_t)B * LI * H);
    float* hstate = allocF((size_t)NL * B * H);
    float* cstate = allocF((size_t)NL * B * H);
    float* gates  = allocF((size_t)B * 4 * H);
    float* proj   = allocF((size_t)B * H);
    float* logits = allocF((size_t)B * LH);
    float* cat_in = allocF((size_t)B * 3 * H);
    float* fusion = allocF((size_t)B * 2 * H);
    float* d1     = allocF((size_t)B * 4 * H);
    float* dec    = allocF((size_t)B * VP);          // padded ldc

    // bf16 staged weights + activations
    bf16* Wih_bf  = allocBF((size_t)NL * 4 * H * H);
    bf16* Whh_bf  = allocBF((size_t)NL * 4 * H * H);
    bf16* Wq_bf   = allocBF((size_t)H * 2 * H);
    bf16* Wh_bf   = allocBF((size_t)H * 2 * H);
    bf16* Wi_bf   = allocBF((size_t)H * 2 * H);
    bf16* Waq_bf  = allocBF((size_t)H * H);
    bf16* Wah_bf  = allocBF((size_t)H * H);
    bf16* Wai_bf  = allocBF((size_t)H * H);
    bf16* Wcat_bf = allocBF((size_t)H * 3 * H);
    bf16* Wd1_bf  = allocBF((size_t)4 * H * 2 * H);
    bf16* Wd2_bf  = allocBF((size_t)VP * 4 * H);     // zero-padded rows
    bf16* emb_bf  = allocBF((size_t)B * T * H);
    bf16* hbf0    = allocBF((size_t)B * H);
    bf16* hbf1    = allocBF((size_t)B * H);
    bf16* cat_bf  = allocBF((size_t)B * 3 * H);
    bf16* fus_bf  = allocBF((size_t)B * 2 * H);
    bf16* d1_bf   = allocBF((size_t)B * 4 * H);

    // ---- one-time staging: weights + emb -> bf16 (Wd2 zero-padded) ----
    launch_cvt(Wih,  Wih_bf,  (size_t)NL * 4 * H * H, stream);
    launch_cvt(Whh,  Whh_bf,  (size_t)NL * 4 * H * H, stream);
    launch_cvt(Wq,   Wq_bf,   (size_t)H * 2 * H, stream);
    launch_cvt(Wh,   Wh_bf,   (size_t)H * 2 * H, stream);
    launch_cvt(Wi,   Wi_bf,   (size_t)H * 2 * H, stream);
    launch_cvt(Wansq,Waq_bf,  (size_t)H * H, stream);
    launch_cvt(Wansh,Wah_bf,  (size_t)H * H, stream);
    launch_cvt(Wansi,Wai_bf,  (size_t)H * H, stream);
    launch_cvt(Wcat, Wcat_bf, (size_t)H * 3 * H, stream);
    launch_cvt(Wd1,  Wd1_bf,  (size_t)4 * H * 2 * H, stream);
    cvt_pad_rows_k<<<4096, 256, 0, stream>>>(Wd2, Wd2_bf, V, VP, 4 * H);
    launch_cvt(emb,  emb_bf,  (size_t)B * T * H, stream);

    // ---- hoisted context projections (f32 A, converted in-kernel; run once) ----
    launch_gemm_f32(question, 2 * H, Wq_bf, bq, nullptr, q_emb, H,
                    B * LQ, H, 2 * H, H, 0, stream);
    launch_gemm_f32(history,  2 * H, Wh_bf, bh, nullptr, h_emb, H,
                    B * LH, H, 2 * H, H, 0, stream);
    launch_gemm_f32(image,    2 * H, Wi_bf, bi, nullptr, i_emb, H,
                    B * LI, H, 2 * H, H, 0, stream);

    // ---- init recurrent state ----
    hipMemcpyAsync(hstate, h0, (size_t)NL * B * H * sizeof(float),
                   hipMemcpyDeviceToDevice, stream);
    hipMemcpyAsync(cstate, c0, (size_t)NL * B * H * sizeof(float),
                   hipMemcpyDeviceToDevice, stream);
    launch_cvt(hstate,         hbf0, (size_t)B * H, stream);
    launch_cvt(hstate + B * H, hbf1, (size_t)B * H, stream);

    const int cellBlocks = (B * H + 255) / 256;

    for (int t = 0; t < T; ++t) {
        // ---------- layer 0 ----------
        launch_gemm_bf(emb_bf + (size_t)t * H, T * H, Wih_bf, bih, bhh,
                       gates, 4 * H, B, 4 * H, H, 4 * H, 0, stream);
        launch_gemm_bf(hbf0, H, Whh_bf, nullptr, nullptr,
                       gates, 4 * H, B, 4 * H, H, 4 * H, 1, stream);   // ACC
        lstm_cell_k<<<cellBlocks, 256, 0, stream>>>(gates, cstate, hstate);
        launch_cvt(hstate, hbf0, (size_t)B * H, stream);

        // ---------- layer 1 ----------
        launch_gemm_bf(hbf0, H, Wih_bf + (size_t)4 * H * H, bih + 4 * H, bhh + 4 * H,
                       gates, 4 * H, B, 4 * H, H, 4 * H, 0, stream);
        launch_gemm_bf(hbf1, H, Whh_bf + (size_t)4 * H * H, nullptr, nullptr,
                       gates, 4 * H, B, 4 * H, H, 4 * H, 1, stream);   // ACC
        lstm_cell_k<<<cellBlocks, 256, 0, stream>>>(gates, cstate + B * H,
                                                    hstate + B * H);
        launch_cvt(hstate + B * H, hbf1, (size_t)B * H, stream);
        const float* out_top = hstate + B * H;                          // [B,H]

        // ---------- attentions: question / history / image ----------
        struct Att { const float* ctx; const bf16* Wans; const float* bans;
                     const float* wa; const float* ba; int Lx; int colOff; };
        const Att atts[3] = {
            { q_emb, Waq_bf, bansq, waq, baq, LQ, 0     },
            { h_emb, Wah_bf, bansh, wah, bah, LH, H     },
            { i_emb, Wai_bf, bansi, wai, bai, LI, 2 * H },
        };
        for (int a = 0; a < 3; ++a) {
            launch_gemm_bf(hbf1, H, atts[a].Wans, atts[a].bans, nullptr,
                           proj, H, B, H, H, H, 0, stream);
            dim3 lg(atts[a].Lx, B);
            attn_logits_k<<<lg, 256, 0, stream>>>(atts[a].ctx, proj, atts[a].wa,
                                                  atts[a].ba, logits, atts[a].Lx);
            attn_combine_k<<<B, 256, 0, stream>>>(atts[a].ctx, logits,
                                                  cat_in + atts[a].colOff, 3 * H,
                                                  atts[a].Lx);
        }

        // ---------- fusion: cat = tanh(cat_in @ Wcat^T + bcat) ----------
        launch_cvt(cat_in, cat_bf, (size_t)B * 3 * H, stream);
        launch_gemm_bf(cat_bf, 3 * H, Wcat_bf, bcat, nullptr,
                       fusion + H, 2 * H, B, H, 3 * H, H, /*tanh*/ 2, stream);
        copy2d_k<<<512, 256, 0, stream>>>(out_top, H, fusion, 2 * H, B, H);

        // ---------- decoder ----------
        launch_cvt(fusion, fus_bf, (size_t)B * 2 * H, stream);
        launch_gemm_bf(fus_bf, 2 * H, Wd1_bf, bd1, nullptr,
                       d1, 4 * H, B, 4 * H, 2 * H, 4 * H, /*relu*/ 4, stream);
        launch_cvt(d1, d1_bf, (size_t)B * 4 * H, stream);
        launch_gemm_bf(d1_bf, 4 * H, Wd2_bf, bd2, nullptr,
                       dec, VP, B, VP, 4 * H, V, 0, stream);

        logsoftmax_k<<<B, 256, 0, stream>>>(dec, VP, out, t);
    }

    // ---- tail outputs: hT, cT ----
    const size_t lps = (size_t)B * T * V;
    hipMemcpyAsync(out + lps, hstate, (size_t)NL * B * H * sizeof(float),
                   hipMemcpyDeviceToDevice, stream);
    hipMemcpyAsync(out + lps + (size_t)NL * B * H, cstate,
                   (size_t)NL * B * H * sizeof(float),
                   hipMemcpyDeviceToDevice, stream);
}
